// LSTMLastOutput_32246614458551
// MI455X (gfx1250) — matrix operand order repaired and checked by reading, not compiled, not run on hardware
//
#include <hip/hip_runtime.h>

typedef __attribute__((ext_vector_type(16))) _Float16 v16h;
typedef __attribute__((ext_vector_type(2)))  __fp16   fp16x2;   // cvt_pkrtz result type
typedef __attribute__((ext_vector_type(8)))  float    v8f;

#define HID    32      // hidden size
#define TLEN   2048    // sequence length
#define BATCH  256
#define ROWS   16      // batch rows per wave (one WMMA M-tile)
#define TCHUNK 32      // x timesteps staged per LDS refill
#define XSTRIDE 20     // padded floats per staged step: 80B (16B aligned, bank-conflict-free)

// Native CDNA5 tanh (V_TANH_F32), confirmed emitted on gfx1250.
#if defined(__has_builtin) && __has_builtin(__builtin_amdgcn_tanhf)
#define FAST_TANH(x) __builtin_amdgcn_tanhf(x)
#else
__device__ __forceinline__ float fast_tanh_fallback(float x) {
    return 2.0f * __builtin_amdgcn_rcpf(1.0f + __builtin_amdgcn_exp2f(x * -2.88539008177792681472f)) - 1.0f;
}
#define FAST_TANH(x) fast_tanh_fallback(x)
#endif

// sigmoid(x) with the 0.5 pre-scale already folded into the gate GEMM:
// acc holds 0.5*x  ->  sigmoid = 0.5*tanh(acc) + 0.5   (1 TRANS + 1 FMA)
__device__ __forceinline__ float sigmoid_prescaled(float half_x) {
    return FAST_TANH(half_x) * 0.5f + 0.5f;
}

__global__ __launch_bounds__(32)
void lstm_last_hidden_wmma(const float* __restrict__ x,
                           const float* __restrict__ W_ih,
                           const float* __restrict__ W_hh,
                           const float* __restrict__ b_ih,
                           const float* __restrict__ b_hh,
                           float* __restrict__ out)
{
    // hbuf uses a K-permuted layout: column c = 2*(j%16) + (j/16), j = hidden index.
    // This makes each lane's (j, j+16) h-pair adjacent -> packed b32 stores.
    // The same permutation is applied to the B operand's K index, so the GEMM
    // result is unchanged (reduction over K is order-invariant).
    __shared__ __align__(16) _Float16 hbuf[ROWS * HID];           // 1 KB
    // x staging, transposed: xs[t][row] (padded) -> per-step rows are contiguous
    __shared__ __align__(16) float    xs[TCHUNK * XSTRIDE];       // 2.5 KB

    const int lane = threadIdx.x;      // 0..31 (wave32)
    const int half = lane >> 4;        // 0: rows 0..7 / K'-low half, 1: rows 8..15 / K'-high half
    const int l16  = lane & 15;
    const int rowBase = blockIdx.x * ROWS;

    // --- Per-lane gate-column constants (N = 16*t + l16 for tile t) ---
    // Tiles 0,1 = i, 2,3 = f, 4,5 = g, 6,7 = o.  i/f/o tiles pre-scaled by 0.5
    // (sigmoid fold); g tiles (4,5) unscaled for plain tanh.
    float wih[8], bias[8];
#pragma unroll
    for (int t = 0; t < 8; ++t) {
        const float sc = (t == 4 || t == 5) ? 1.0f : 0.5f;
        const int g = t * 16 + l16;
        wih[t]  = W_ih[g] * sc;                  // W_ih is [128,1]
        bias[t] = (b_ih[g] + b_hh[g]) * sc;
    }

    // --- Load W_hh once into WMMA B layout (f16), K-permuted + gate-scaled ---
    // B'[K'=c][N=g] = sc * W_hh[g][ j(c) ],  j(c) = (c&1)*16 + (c>>1).
    // Lane covers K' = half*8 + k (k<8), 16 + half*8 + (k-8) (k>=8); N = 16t + l16.
    v16h Bm[8];
#pragma unroll
    for (int t = 0; t < 8; ++t) {
        const float sc = (t == 4 || t == 5) ? 1.0f : 0.5f;
        const float* wrow = W_hh + (t * 16 + l16) * HID;
#pragma unroll
        for (int k = 0; k < 16; ++k) {
            const int c = (k & 8) * 2 + half * 8 + (k & 7);
            const int j = (c & 1) * 16 + (c >> 1);
            Bm[t][k] = (_Float16)(wrow[j] * sc);
        }
    }

    // --- c state in C/D layout: VGPR v <-> row M = half*8+v; tiles 0/1 cover j=0..15 / 16..31 ---
    v8f c0v = {}, c1v = {};

    // h starts at zero: clear hbuf (512 halves, 16 per lane)
#pragma unroll
    for (int i = 0; i < ROWS * HID / 32; ++i) hbuf[lane + 32 * i] = (_Float16)0.0f;
    __syncthreads();   // single-wave workgroup: S_NOP in HW

    const float* xrow = x + (size_t)rowBase * TLEN;

    for (int tc = 0; tc < TLEN; tc += TCHUNK) {
        // Stage 32 timesteps of all 16 rows. Lane `lane` owns step (tc+lane):
        // coalesced global loads, then 4x ds_store_b128 into xs[lane][0..15].
        float xv[ROWS];
#pragma unroll
        for (int r = 0; r < ROWS; ++r)
            xv[r] = xrow[(size_t)r * TLEN + tc + lane];
        {
            float4* dst = (float4*)(xs + lane * XSTRIDE);
            dst[0] = make_float4(xv[0],  xv[1],  xv[2],  xv[3]);
            dst[1] = make_float4(xv[4],  xv[5],  xv[6],  xv[7]);
            dst[2] = make_float4(xv[8],  xv[9],  xv[10], xv[11]);
            dst[3] = make_float4(xv[12], xv[13], xv[14], xv[15]);
        }
        __syncthreads();

        for (int s = 0; s < TCHUNK; ++s) {
            // --- A operand (previous h) from LDS in WMMA A layout (K' space) ---
            // lane row M = l16; half==0 -> K' {0..7, 16..23}; half==1 -> K' {8..15, 24..31}
            const _Float16* hrow = hbuf + l16 * HID + half * 8;
            v16h A;
#pragma unroll
            for (int k = 0; k < 8; ++k) A[k] = hrow[k];
#pragma unroll
            for (int k = 0; k < 8; ++k) A[8 + k] = hrow[16 + k];

            // --- This step's x values for the lane's 8 rows: 2x ds_load_b128 broadcast ---
            const float* xsrow = xs + s * XSTRIDE + half * 8;
            const float4 xa = *(const float4*)(xsrow);
            const float4 xb = *(const float4*)(xsrow + 4);
            const float xr[8] = { xa.x, xa.y, xa.z, xa.w, xb.x, xb.y, xb.z, xb.w };

            // --- C init: xg = x*W_ih + (b_ih+b_hh) (gate-scaled), directly in C layout ---
            v8f acc[8];
#pragma unroll
            for (int t = 0; t < 8; ++t) {
#pragma unroll
                for (int v = 0; v < 8; ++v)
                    acc[t][v] = xr[v] * wih[t] + bias[t];
            }

            // --- Recurrent GEMM: 8 independent 16x16x32 f16 WMMAs ---
#pragma unroll
            for (int t = 0; t < 8; ++t)
                acc[t] = __builtin_amdgcn_wmma_f32_16x16x32_f16(
                    /*neg_a=*/false, A, /*neg_b=*/false, Bm[t],
                    /*c_mod=*/(short)0, acc[t], /*reuse_a=*/false, /*reuse_b=*/false);

            // --- Elementwise LSTM cell; i/f/o accs are pre-scaled by 0.5 ---
#pragma unroll
            for (int v = 0; v < 8; ++v) {
                const int row = half * 8 + v;
                // tile 0 set: hidden j = l16
                const float i0 = sigmoid_prescaled(acc[0][v]);
                const float f0 = sigmoid_prescaled(acc[2][v]);
                const float g0 = FAST_TANH        (acc[4][v]);
                const float o0 = sigmoid_prescaled(acc[6][v]);
                const float cc0 = f0 * c0v[v] + i0 * g0;
                c0v[v] = cc0;
                const float h0 = o0 * FAST_TANH(cc0);
                // tile 1 set: hidden j = 16 + l16
                const float i1 = sigmoid_prescaled(acc[1][v]);
                const float f1 = sigmoid_prescaled(acc[3][v]);
                const float g1 = FAST_TANH        (acc[5][v]);
                const float o1 = sigmoid_prescaled(acc[7][v]);
                const float cc1 = f1 * c1v[v] + i1 * g1;
                c1v[v] = cc1;
                const float h1 = o1 * FAST_TANH(cc1);
                // Packed f16 store: cols (2*l16, 2*l16+1) = hidden (l16, 16+l16)
                const fp16x2 p = __builtin_amdgcn_cvt_pkrtz(h0, h1);
                *(fp16x2*)(hbuf + row * HID + l16 * 2) = p;
            }
            // No barrier: single wave; DS ops in-order, compiler preserves
            // hbuf store->load order (may-alias).
        }
        __syncthreads();   // xs chunk reuse fence (S_NOP in HW)
    }

    // --- Emit last hidden state: out[(rowBase+row)*32 + j], K-permuted read ---
#pragma unroll
    for (int v = 0; v < 8; ++v) {
        const int row = half * 8 + v;
        out[(size_t)(rowBase + row) * HID + l16]      = (float)hbuf[row * HID + l16 * 2];
        out[(size_t)(rowBase + row) * HID + 16 + l16] = (float)hbuf[row * HID + l16 * 2 + 1];
    }
}

extern "C" void kernel_launch(void* const* d_in, const int* in_sizes, int n_in,
                              void* d_out, int out_size, void* d_ws, size_t ws_size,
                              hipStream_t stream) {
    (void)in_sizes; (void)n_in; (void)d_ws; (void)ws_size; (void)out_size;
    const float* x    = (const float*)d_in[0];   // [256, 2048, 1]
    const float* W_ih = (const float*)d_in[1];   // [128, 1]
    const float* W_hh = (const float*)d_in[2];   // [128, 32]
    const float* b_ih = (const float*)d_in[3];   // [128]
    const float* b_hh = (const float*)d_in[4];   // [128]
    float* out = (float*)d_out;                  // [256, 32]

    dim3 grid(BATCH / ROWS);   // 16 independent single-wave workgroups
    dim3 block(32);            // one wave32 per workgroup
    lstm_last_hidden_wmma<<<grid, block, 0, stream>>>(x, W_ih, W_hh, b_ih, b_hh, out);
}
